// MyGCN_29386166239367
// MI455X (gfx1250) — matrix operand order repaired and checked by reading, not compiled, not run on hardware
//
#include <hip/hip_runtime.h>

typedef __attribute__((ext_vector_type(16))) __bf16 v16bf;
typedef __attribute__((ext_vector_type(4)))  __bf16 v4bf;
typedef __attribute__((ext_vector_type(8)))  float  v8f;

#define K_DIM 128

// ---------------- degree / norm kernels ----------------

__global__ void gcn_deg_init(float* __restrict__ deg, int n) {
  int i = blockIdx.x * blockDim.x + threadIdx.x;
  if (i < n) deg[i] = 1.0f;  // self-loop contributes 1
}

__global__ void gcn_deg_count(const int* __restrict__ ei, float* __restrict__ deg, int ne) {
  int e = blockIdx.x * blockDim.x + threadIdx.x;
  if (e < ne) atomicAdd(&deg[ei[ne + e]], 1.0f);  // dst = second row of edge_index
}

__global__ void gcn_dinv(const float* __restrict__ deg, float* __restrict__ dinv, int n) {
  int i = blockIdx.x * blockDim.x + threadIdx.x;
  if (i < n) {
    float d = deg[i];
    dinv[i] = (d > 0.0f) ? rsqrtf(d) : 0.0f;
  }
}

// ---------------- pack W (f32 row-major [128][fout]) into bf16 B-fragment order ----------------
// B fragment (32x16 tile at K-tile kt, N-tile nt): lane = col%16 + 16*(K>=16 half), elem e = K%16.
// Wp layout: [kt][nt][lane][e] -> each lane's 16 bf16 are 32 contiguous bytes.

__global__ void gcn_pack_w(const float* __restrict__ W, __bf16* __restrict__ Wp,
                           int fout, int foutShift) {
  int idx = blockIdx.x * blockDim.x + threadIdx.x;  // over 128*fout
  if (idx >= (K_DIM << foutShift)) return;
  int k = idx >> foutShift;
  int n = idx & (fout - 1);
  int kt = k >> 5, rel = k & 31;
  int hf = rel >> 4, e = rel & 15;
  int nt = n >> 4, ncol = n & 15;
  int ntn = fout >> 4;
  size_t dst = ((size_t)(kt * ntn + nt) * 32 + (ncol + 16 * hf)) * 16 + e;
  Wp[dst] = (__bf16)W[idx];
}

// ---------------- WMMA GEMM: out[nrows][fout] = relu?(A)[nrows][128] @ W[128][fout] --------------
// Block covers 64 rows x fout cols; fout/16 waves; each wave owns one 16-col N-tile and loops
// 4 M-tiles, reusing each B fragment across 4 v_wmma_f32_16x16x32_bf16 (16 WMMAs per wave).
// A tile staged in LDS in bf16 fragment order: [kk][mt][lane][e] (16 KB).

__global__ __launch_bounds__(256) void gcn_gemm_wmma(
    const float* __restrict__ A, const __bf16* __restrict__ Wp, float* __restrict__ out,
    int nrows, int fout, int applyRelu)
{
  __shared__ __bf16 lds_a[4 * 4 * 32 * 16];  // [kk][mt][lane][e], 16 KB

  const int tid   = threadIdx.x;
  const int nthr  = blockDim.x;
  const int tileM = blockIdx.x * 64;

  // Stage + convert 64x128 A tile into fragment order.
  // A-fragment map: rel=k%32; hf=rel>>4, e=rel&15.
  // k aligned to 4 => the 4 elements land at e, e+1, e+2, e+3 in one lane: one b64 store.
  for (int idx = tid * 4; idx < 64 * K_DIM; idx += nthr * 4) {
    int m = idx >> 7;
    int k = idx & (K_DIM - 1);
    int grow = tileM + m;
    float4 v = make_float4(0.f, 0.f, 0.f, 0.f);
    if (grow < nrows) {
      v = *reinterpret_cast<const float4*>(A + (size_t)grow * K_DIM + k);
      if (applyRelu) {
        v.x = fmaxf(v.x, 0.f); v.y = fmaxf(v.y, 0.f);
        v.z = fmaxf(v.z, 0.f); v.w = fmaxf(v.w, 0.f);
      }
    }
    int kk = k >> 5, rel = k & 31;
    int hf, e;
    hf = rel >> 4; e = rel & 15;
    int lane = (m & 15) + 16 * hf;
    int mt = m >> 4;
    v4bf p;
    p.x = (__bf16)v.x; p.y = (__bf16)v.y; p.z = (__bf16)v.z; p.w = (__bf16)v.w;
    *reinterpret_cast<v4bf*>(&lds_a[(((kk * 4 + mt) * 32 + lane) << 4) + e]) = p;
  }
  __syncthreads();

  const int wave  = tid >> 5;         // = N-tile index (fout/16 waves)
  const int lane  = tid & 31;
  const int ntn   = fout >> 4;

  v8f acc[4] = {};
  #pragma unroll
  for (int kk = 0; kk < 4; ++kk) {
    // one 32-byte B-fragment load, reused across 4 M-tiles
    v16bf b = *reinterpret_cast<const v16bf*>(
        Wp + ((size_t)(kk * ntn + wave) * 32 + lane) * 16);
    #pragma unroll
    for (int mt = 0; mt < 4; ++mt) {
      v16bf a = *reinterpret_cast<const v16bf*>(
          &lds_a[((kk * 4 + mt) * 32 + lane) << 4]);
      acc[mt] = __builtin_amdgcn_wmma_f32_16x16x32_bf16(
          false, a, false, b, (short)0, acc[mt], false, false);
    }
  }

  // D layout: VGPR r -> M = r + 8*(lane>=16), N = lane%16
  const int hf   = lane >> 4;
  const int ncol = lane & 15;
  const int tileN = wave * 16;
  #pragma unroll
  for (int mt = 0; mt < 4; ++mt) {
    #pragma unroll
    for (int r = 0; r < 8; ++r) {
      int m = tileM + mt * 16 + r + hf * 8;
      if (m < nrows) out[(size_t)m * fout + tileN + ncol] = acc[mt][r];
    }
  }
}

// ---------------- self-loop + bias init: out = b + xw * dinv^2 ----------------

__global__ void gcn_init_out(const float* __restrict__ xw, const float* __restrict__ bias,
                             const float* __restrict__ dinv, float* __restrict__ out,
                             int nrows, int fout)
{
  int i = blockIdx.x * blockDim.x + threadIdx.x;
  if (i >= nrows) return;
  int c = blockIdx.y * 4;
  float di = dinv[i];
  float sl = di * di;
  float4 v = *reinterpret_cast<const float4*>(xw + (size_t)i * fout + c);
  float4 b = *reinterpret_cast<const float4*>(bias + c);
  float4 o;
  o.x = b.x + v.x * sl; o.y = b.y + v.y * sl;
  o.z = b.z + v.z * sl; o.w = b.w + v.w * sl;
  *reinterpret_cast<float4*>(out + (size_t)i * fout + c) = o;
}

// ---------------- edge aggregation: out[dst] += xw[src] * dinv[src]*dinv[dst] ----------------
// grid.y = fout/4 channel groups, grid.x over edges. Gather + L2-resident f32 atomics.

__global__ void gcn_edge_agg(const int* __restrict__ ei, const float* __restrict__ xw,
                             const float* __restrict__ dinv, float* __restrict__ out,
                             int ne, int fout)
{
  int e = blockIdx.x * blockDim.x + threadIdx.x;
  if (e >= ne) return;
  int c = blockIdx.y * 4;
  int s = ei[e];
  int d = ei[ne + e];
  float norm = dinv[s] * dinv[d];
  float4 v = *reinterpret_cast<const float4*>(xw + (size_t)s * fout + c);
  float* o = out + (size_t)d * fout + c;
  atomicAdd(o + 0, v.x * norm);
  atomicAdd(o + 1, v.y * norm);
  atomicAdd(o + 2, v.z * norm);
  atomicAdd(o + 3, v.w * norm);
}

// ---------------- launch ----------------

static inline int ceil_div(int a, int b) { return (a + b - 1) / b; }

extern "C" void kernel_launch(void* const* d_in, const int* in_sizes, int n_in,
                              void* d_out, int out_size, void* d_ws, size_t ws_size,
                              hipStream_t stream) {
  const float* x  = (const float*)d_in[0];
  const int*   ei = (const int*)d_in[1];
  const float* W1 = (const float*)d_in[2];
  const float* b1 = (const float*)d_in[3];
  const float* W2 = (const float*)d_in[4];
  const float* b2 = (const float*)d_in[5];
  const float* W3 = (const float*)d_in[6];
  const float* b3 = (const float*)d_in[7];
  float* outp = (float*)d_out;

  const int N = in_sizes[0] / 128;   // 50000
  const int E = in_sizes[1] / 2;     // 800000

  // workspace carve-up (all L2-resident: ~52 MB vs 192 MB L2)
  float*  deg  = (float*)d_ws;
  float*  dinv = deg + N;
  float*  xw   = dinv + N;                         // N*128 GEMM scratch
  float*  h    = xw + (size_t)N * 128;             // N*128 hidden activations (pre-ReLU)
  __bf16* Wp1  = (__bf16*)(h + (size_t)N * 128);   // packed bf16 weights
  __bf16* Wp2  = Wp1 + 128 * 128;
  __bf16* Wp3  = Wp2 + 128 * 128;

  const int T = 256;

  // one-time: degrees (with self-loops), symmetric norm, packed bf16 weights
  gcn_deg_init <<<ceil_div(N, T), T, 0, stream>>>(deg, N);
  gcn_deg_count<<<ceil_div(E, T), T, 0, stream>>>(ei, deg, E);
  gcn_dinv     <<<ceil_div(N, T), T, 0, stream>>>(deg, dinv, N);
  gcn_pack_w   <<<ceil_div(128 * 128, T), T, 0, stream>>>(W1, Wp1, 128, 7);
  gcn_pack_w   <<<ceil_div(128 * 128, T), T, 0, stream>>>(W2, Wp2, 128, 7);
  gcn_pack_w   <<<ceil_div(128 *  64, T), T, 0, stream>>>(W3, Wp3,  64, 6);

  const int mblocks = ceil_div(N, 64);

  // ---- layer 1: 128 -> 128 ----
  gcn_gemm_wmma<<<mblocks, 256, 0, stream>>>(x, Wp1, xw, N, 128, /*relu in*/0);
  gcn_init_out <<<dim3(ceil_div(N, T), 32), T, 0, stream>>>(xw, b1, dinv, h, N, 128);
  gcn_edge_agg <<<dim3(ceil_div(E, T), 32), T, 0, stream>>>(ei, xw, dinv, h, E, 128);

  // ---- layer 2: 128 -> 128 (ReLU fused into A-load) ----
  gcn_gemm_wmma<<<mblocks, 256, 0, stream>>>(h, Wp2, xw, N, 128, /*relu in*/1);
  gcn_init_out <<<dim3(ceil_div(N, T), 32), T, 0, stream>>>(xw, b2, dinv, h, N, 128);
  gcn_edge_agg <<<dim3(ceil_div(E, T), 32), T, 0, stream>>>(ei, xw, dinv, h, E, 128);

  // ---- layer 3: 128 -> 64 (ReLU fused into A-load), writes d_out ----
  gcn_gemm_wmma<<<mblocks, 128, 0, stream>>>(h, Wp3, xw, N, 64, /*relu in*/1);
  gcn_init_out <<<dim3(ceil_div(N, T), 16), T, 0, stream>>>(xw, b3, dinv, outp, N, 64);
  gcn_edge_agg <<<dim3(ceil_div(E, T), 16), T, 0, stream>>>(ei, xw, dinv, outp, E, 64);
}
